// Attention_72653666779808
// MI455X (gfx1250) — matrix operand order repaired and checked by reading, not compiled, not run on hardware
//
#include <hip/hip_runtime.h>
#include <hip/hip_bf16.h>

// ---------------------------------------------------------------------------
// Problem constants (from the reference)
// ---------------------------------------------------------------------------
#define B_    2
#define T_    2048
#define D_    2048
#define H_    16
#define KV_   4
#define HD_   128
#define M_    (B_ * T_)        // 4096 rows of x
#define NQKV_ (D_ + 2 * KV_ * HD_)   // 3072 = Q(2048) + K(512) + V(512)

#define QSCALE   0.08838834764831845f   // HD^-0.5
#define LOG10000 9.210340371976184f

typedef __attribute__((ext_vector_type(16))) _Float16 v16h;
typedef __attribute__((ext_vector_type(8)))  _Float16 v8h;
typedef __attribute__((ext_vector_type(4)))  _Float16 v4h;
typedef __attribute__((ext_vector_type(8)))  float    v8f;

__device__ __forceinline__ v8h  ld8 (const _Float16* p) { return *(const v8h *)p; }
__device__ __forceinline__ v16h ld16(const _Float16* p) { return *(const v16h*)p; }
__device__ __forceinline__ v16h cat8(v8h lo, v8h hi) {
  return __builtin_shufflevector(lo, hi, 0,1,2,3,4,5,6,7,8,9,10,11,12,13,14,15);
}
// D = A(16x32 f16) * B(32x16 f16) + C(16x16 f32), wave32 CDNA5 WMMA
__device__ __forceinline__ v8f wmma_f16(v16h a, v16h b, v8f c) {
  return __builtin_amdgcn_wmma_f32_16x16x32_f16(false, a, false, b, (short)0, c,
                                                false, false);
}
__device__ __forceinline__ void sched_fence() {
#if __has_builtin(__builtin_amdgcn_sched_barrier)
  __builtin_amdgcn_sched_barrier(0);   // no instruction; scheduling fence only
#endif
}

// ---------------------------------------------------------------------------
// fp32 -> f16 cast, 4 elements / thread
// ---------------------------------------------------------------------------
__global__ void cast_f32_f16_x4(const float* __restrict__ in,
                                _Float16* __restrict__ out, int n4) {
  int i = blockIdx.x * blockDim.x + threadIdx.x;
  if (i >= n4) return;
  float4 v = ((const float4*)in)[i];
  v4h o = { (_Float16)v.x, (_Float16)v.y, (_Float16)v.z, (_Float16)v.w };
  ((v4h*)out)[i] = o;
}

// ---------------------------------------------------------------------------
// C(MxN) = A(MxK, f16 row-major) * Bw(NxK, f16 row-major)^T
// block = 128 threads (4 waves along N); each wave computes a 64x64 tile as a
// 4x4 grid of v_wmma_f32_16x16x32_f16, K swept in steps of 32.
// 16 WMMAs per 8KB of (L2-resident) loads -> ~33 FLOP/B from cache.
// ---------------------------------------------------------------------------
template <typename OutT>
__global__ __launch_bounds__(128) void gemm_a_bT(
    const _Float16* __restrict__ A, const _Float16* __restrict__ Bw,
    OutT* __restrict__ C, int M, int N, int K) {
  const int lane = threadIdx.x & 31;
  const int wave = threadIdx.x >> 5;
  const int hh   = lane >> 4;      // half-wave
  const int l16  = lane & 15;
  const int row0 = blockIdx.y * 64;
  const int col0 = blockIdx.x * 256 + wave * 64;

  const v8f zero8 = {0.f,0.f,0.f,0.f,0.f,0.f,0.f,0.f};
  v8f acc[4][4];
#pragma unroll
  for (int i = 0; i < 4; ++i)
#pragma unroll
    for (int j = 0; j < 4; ++j) acc[i][j] = zero8;

  for (int k0 = 0; k0 < K; k0 += 32) {
    v16h afr[4];
#pragma unroll
    for (int mi = 0; mi < 4; ++mi) {
      // A layout: lane holds row m, contiguous K (h=0: 0..15, h=1: 16..31)
      const _Float16* p = A + (size_t)(row0 + mi * 16 + l16) * K + k0 + hh * 16;
      afr[mi] = ld16(p);
    }
    v16h bfr[4];
#pragma unroll
    for (int ni = 0; ni < 4; ++ni) {
      // B layout: lane holds column n, contiguous K (h=0: 0..15, h=1: 16..31)
      const _Float16* p = Bw + (size_t)(col0 + ni * 16 + l16) * K + k0 + hh * 16;
      bfr[ni] = ld16(p);
    }
#pragma unroll
    for (int mi = 0; mi < 4; ++mi)
#pragma unroll
      for (int ni = 0; ni < 4; ++ni)
        acc[mi][ni] = wmma_f16(afr[mi], bfr[ni], acc[mi][ni]);
  }

  // C/D layout: VGPR r, lanes 0-15 -> M=r, lanes 16-31 -> M=r+8; N = lane&15
#pragma unroll
  for (int mi = 0; mi < 4; ++mi)
#pragma unroll
    for (int ni = 0; ni < 4; ++ni) {
      const int n  = col0 + ni * 16 + l16;
      const int mb = row0 + mi * 16 + 8 * hh;
#pragma unroll
      for (int r = 0; r < 8; ++r)
        C[(size_t)(mb + r) * N + n] = (OutT)acc[mi][ni][r];
    }
}

// ---------------------------------------------------------------------------
// RoPE + scatter. Reads fused qkv (M x 3072, f16), applies rotation in f32,
// folds HD^-0.5 into Q, writes:
//   Qo: (B,H,T,HD)  Ko: (B,KV,T,HD)  Vt: (B,KV,HD,T)  (V transposed!)
// grid = (7, M_), block = 256  -> j in [0,1792): 1024 Q-pairs, 256 K-pairs,
// 512 V elements per token row.
// ---------------------------------------------------------------------------
__global__ void rope_scatter(const _Float16* __restrict__ qkv,
                             _Float16* __restrict__ Qo,
                             _Float16* __restrict__ Ko,
                             _Float16* __restrict__ Vt) {
  const int m = blockIdx.y;
  const int j = blockIdx.x * blockDim.x + threadIdx.x;
  const int b = m / T_, t = m % T_;
  const _Float16* row = qkv + (size_t)m * NQKV_;

  if (j < 1024) {                       // Q: head h, rotation pair d / d+64
    const int h = j >> 6, d = j & 63;
    float v0 = (float)row[h * HD_ + d];
    float v1 = (float)row[h * HD_ + d + 64];
    float ang = (float)t * __expf((float)d * (-LOG10000 / 64.f));
    float s, c; __sincosf(ang, &s, &c);
    _Float16* q = Qo + ((size_t)(b * H_ + h) * T_ + t) * HD_;
    q[d]      = (_Float16)((v0 * c - v1 * s) * QSCALE);
    q[d + 64] = (_Float16)((v1 * c + v0 * s) * QSCALE);
  } else if (j < 1280) {                // K: kv-head g
    const int jj = j - 1024, g = jj >> 6, d = jj & 63;
    float v0 = (float)row[D_ + g * HD_ + d];
    float v1 = (float)row[D_ + g * HD_ + d + 64];
    float ang = (float)t * __expf((float)d * (-LOG10000 / 64.f));
    float s, c; __sincosf(ang, &s, &c);
    _Float16* k = Ko + ((size_t)(b * KV_ + g) * T_ + t) * HD_;
    k[d]      = (_Float16)(v0 * c - v1 * s);
    k[d + 64] = (_Float16)(v1 * c + v0 * s);
  } else if (j < 1792) {                // V: transpose to (hd, t)
    const int jj = j - 1280, g = jj >> 7, d = jj & 127;
    Vt[((size_t)(b * KV_ + g) * HD_ + d) * T_ + t] = row[D_ + KV_ * HD_ + g * HD_ + d];
  }
}

// ---------------------------------------------------------------------------
// Flash attention, transposed formulation (queries live in lanes).
// One wave owns 16 queries of one (b,h); loops keys in tiles of 32.
//   S^T = K_tile (16x128) @ Q^T       -> queries are WMMA columns (lanes)
//   O^T = V^T   (16x32 keys) @ P^T    -> P^T rebuilt from S^T C-layout with
//                                        shfl_xor(16) only — no LDS.
// Per tile: issue ALL K then V loads (pinned by a scheduling fence). Loads
// retire in order, so the S WMMAs wait only on the K portion while the V
// loads stream behind; the softmax VALU/transcendental work covers the tail,
// and the PV WMMAs consume fully prefetched fragments.
// block = 128 (4 waves = 4 q-tiles), grid = (T/64, H, B).
// ---------------------------------------------------------------------------
__global__ __launch_bounds__(128) void attn_fwd(
    const _Float16* __restrict__ Q, const _Float16* __restrict__ Kr,
    const _Float16* __restrict__ Vt, _Float16* __restrict__ Ctx) {
  const int lane = threadIdx.x & 31;
  const int wave = threadIdx.x >> 5;
  const int hh   = lane >> 4;
  const int l16  = lane & 15;
  const int qt   = blockIdx.x * 4 + wave;    // q tile of 16
  const int h    = blockIdx.y, b = blockIdx.z;
  const int g    = h >> 2;                   // GQA group (H/KV = 4)

  const _Float16* Qb = Q  + ((size_t)(b * H_  + h) * T_)  * HD_;
  const _Float16* Kb = Kr + ((size_t)(b * KV_ + g) * T_)  * HD_;
  const _Float16* Vb = Vt + ((size_t)(b * KV_ + g) * HD_) * T_;

  // Resident Q B-fragments: lane's column q = qt*16 + l16, 4 chunks of K(d)=32
  const int q = qt * 16 + l16;
  v16h qf[4];
#pragma unroll
  for (int c = 0; c < 4; ++c)
    qf[c] = cat8(ld8(Qb + (size_t)q * HD_ + c * 32 + hh * 8), ld8(Qb + (size_t)q * HD_ + c * 32 + hh * 8 + 16));

  const v8f zero8 = {0.f,0.f,0.f,0.f,0.f,0.f,0.f,0.f};
  v8f o[8];                                   // O^T: 8 d-tiles of 16
#pragma unroll
  for (int i = 0; i < 8; ++i) o[i] = zero8;
  float m = -1e30f, l = 0.f;

  for (int kt = 0; kt < T_; kt += 32) {
    // ---- issue K-tile loads first (A-layout: rows = keys, K-dim = d) ----
    v16h kf[8];
#pragma unroll
    for (int c = 0; c < 4; ++c) {
      const _Float16* kp = Kb + (size_t)(kt + l16) * HD_ + c * 32 + hh * 8;
      kf[2 * c]     = cat8(ld8(kp), ld8(kp + 16));
      const _Float16* kp1 = kp + 16 * HD_;
      kf[2 * c + 1] = cat8(ld8(kp1), ld8(kp1 + 16));
    }
    // ---- then issue V-tile loads (retire after K; hidden by softmax) ----
    v16h vf[8];
#pragma unroll
    for (int dt = 0; dt < 8; ++dt) {
      const _Float16* vp = Vb + (size_t)(dt * 16 + l16) * T_ + kt + hh * 16;
      vf[dt] = ld16(vp);
    }
    // Pin issue order: all 24 loads must be issued before any compute below
    // is scheduled; S WMMAs then use partial loadcnt waits on the K portion.
    sched_fence();

    // ---- S^T = K @ Q^T : two 16(key)x16(q) C-tiles, K(d) swept 4x32 ----
    v8f s0 = zero8, s1 = zero8;
#pragma unroll
    for (int c = 0; c < 4; ++c) {
      s0 = wmma_f16(kf[2 * c],     qf[c], s0);
      s1 = wmma_f16(kf[2 * c + 1], qf[c], s1);
    }
    // ---- online softmax over this lane's query column ----
    float mx = -1e30f;
#pragma unroll
    for (int r = 0; r < 8; ++r) mx = fmaxf(mx, fmaxf(s0[r], s1[r]));
    mx = fmaxf(mx, __shfl_xor(mx, 16));       // other half holds other key rows
    const float m_new = fmaxf(m, mx);
    const float corr  = __expf(m - m_new);
    float p0[8], p1[8], ps = 0.f;
#pragma unroll
    for (int r = 0; r < 8; ++r) {
      p0[r] = __expf(s0[r] - m_new);
      p1[r] = __expf(s1[r] - m_new);
      ps += p0[r] + p1[r];
    }
    ps += __shfl_xor(ps, 16);
    l = l * corr + ps;
    m = m_new;
#pragma unroll
    for (int i = 0; i < 8; ++i) o[i] *= corr;
    // ---- build P^T B-fragment (K=32 keys, N=16 queries) from C-layout ----
    // h=0 lane needs keys 0..15: own p0 + partner(p0); h=1: partner(p1) + own p1
    v16h pB;
#pragma unroll
    for (int r = 0; r < 8; ++r) {
      float x0 = __shfl_xor(p0[r], 16);
      float x1 = __shfl_xor(p1[r], 16);
      float lo = hh ? x1    : p0[r];
      float hi = hh ? p1[r] : x0;
      pB[r]     = (_Float16)lo;
      pB[r + 8] = (_Float16)hi;
    }
    // ---- O^T += V^T @ P^T : 8 d-tiles, V fragments already prefetched ----
#pragma unroll
    for (int dt = 0; dt < 8; ++dt)
      o[dt] = wmma_f16(vf[dt], pB, o[dt]);
  }

  // ---- normalize and write Ctx (B, T, H*HD) ----
  const float invl = 1.f / l;
#pragma unroll
  for (int dt = 0; dt < 8; ++dt) {
    v8h outv;
#pragma unroll
    for (int r = 0; r < 8; ++r) outv[r] = (_Float16)(o[dt][r] * invl);
    _Float16* cp = Ctx + ((size_t)(b * T_ + qt * 16 + l16)) * D_
                 + h * HD_ + dt * 16 + 8 * hh;
    *(v8h*)cp = outv;
  }
}

// ---------------------------------------------------------------------------
// Host launcher
// ---------------------------------------------------------------------------
extern "C" void kernel_launch(void* const* d_in, const int* in_sizes, int n_in,
                              void* d_out, int out_size, void* d_ws, size_t ws_size,
                              hipStream_t stream) {
  const float* x  = (const float*)d_in[0];
  const float* wq = (const float*)d_in[1];
  const float* wk = (const float*)d_in[2];
  const float* wv = (const float*)d_in[3];
  const float* wo = (const float*)d_in[4];
  float* out = (float*)d_out;

  char* p = (char*)d_ws;
  _Float16* x_h  = (_Float16*)p; p += (size_t)M_ * D_ * 2;        // 16 MB
  _Float16* wqkv = (_Float16*)p; p += (size_t)NQKV_ * D_ * 2;     // 12 MB
  _Float16* wo_h = (_Float16*)p; p += (size_t)D_ * D_ * 2;        //  8 MB
  _Float16* qkv  = (_Float16*)p; p += (size_t)M_ * NQKV_ * 2;     // 24 MB
  _Float16* q_r  = (_Float16*)p; p += (size_t)B_ * H_  * T_ * HD_ * 2;  // 16 MB
  _Float16* k_r  = (_Float16*)p; p += (size_t)B_ * KV_ * T_ * HD_ * 2;  //  4 MB
  _Float16* v_t  = (_Float16*)p; p += (size_t)B_ * KV_ * T_ * HD_ * 2;  //  4 MB
  _Float16* ctx  = (_Float16*)p; p += (size_t)M_ * D_ * 2;        // 16 MB

  auto cast = [&](const float* src, _Float16* dst, size_t n) {
    int n4 = (int)(n / 4);
    cast_f32_f16_x4<<<dim3((n4 + 255) / 256), dim3(256), 0, stream>>>(src, dst, n4);
  };
  cast(x,  x_h, (size_t)M_ * D_);
  cast(wq, wqkv, (size_t)D_ * D_);                                       // rows 0..2047
  cast(wk, wqkv + (size_t)D_ * D_, (size_t)(KV_ * HD_) * D_);            // rows 2048..2559
  cast(wv, wqkv + (size_t)(D_ + KV_ * HD_) * D_, (size_t)(KV_ * HD_) * D_); // rows 2560..3071
  cast(wo, wo_h, (size_t)D_ * D_);

  // Fused QKV projection: (4096 x 2048) @ (3072 x 2048)^T -> f16
  gemm_a_bT<_Float16><<<dim3(NQKV_ / 256, M_ / 64), dim3(128), 0, stream>>>(
      x_h, wqkv, qkv, M_, NQKV_, D_);

  // RoPE (+ q scale) and layout scatter (V transposed)
  rope_scatter<<<dim3(7, M_), dim3(256), 0, stream>>>(qkv, q_r, k_r, v_t);

  // Flash attention: 4096 independent waves
  attn_fwd<<<dim3(T_ / 64, H_, B_), dim3(128), 0, stream>>>(q_r, k_r, v_t, ctx);

  // Output projection: (4096 x 2048) @ (2048 x 2048)^T -> f32 out
  gemm_a_bT<float><<<dim3(D_ / 256, M_ / 64), dim3(128), 0, stream>>>(
      ctx, wo_h, out, M_, D_, D_);
}
